// CSTRN_27522150433175
// MI455X (gfx1250) — matrix operand order repaired and checked by reading, not compiled, not run on hardware
//
#include <hip/hip_runtime.h>
#include <hip/hip_bf16.h>
#include <math.h>

// ---------------------------------------------------------------------------
// MI455X (gfx1250) fused attention-renderer kernel.
// wave32, WMMA f32_16x16x32_f16, one workgroup = 8 waves = 128 query rows.
// All per-row intermediates are wave-private (rows 16w..16w+15 belong to the
// threads of wave w), so the n-loop needs only 3 workgroup barriers.
// ---------------------------------------------------------------------------

typedef _Float16 v16h __attribute__((ext_vector_type(16)));
typedef _Float16 v8h  __attribute__((ext_vector_type(8)));
typedef float    v8f  __attribute__((ext_vector_type(8)));

#define EMB   32
#define CELL  128
#define QTOT  1536       // D*H*W
#define BB    32
#define NN    8
#define KEYS  256        // H*W
#define QTILE 128        // queries per workgroup
#define NQT   12         // QTOT / QTILE

// LDS strides (in halves). All multiples of 8 halves (16B) so every fragment
// load is a ds_load_b128; bank steps (stride/2 mod 64) = 20 / 4 / 36 -> the
// 16 lanes of a fragment hit distinct banks (conflict-free).
#define S_INP 40
#define S_W1  40
#define S_EO  40
#define S_A1  40
#define S_A2  72
#define S_W2  264
#define S_ATT 264
#define S_V   264

#define OUT_ELEMS   (BB * CELL * QTOT)         // 6291456
#define QUAT_O_OFF  OUT_ELEMS                  // pose_o_quat [B,N,7]
#define QUAT_Q_OFF  (OUT_ELEMS + BB * NN * 7)  // pose_q_quat [B,1,7]

struct __align__(16) Smem {
    _Float16 wq1t[256 * S_W1];    // Wq1^T  [j=256][k], k>=15 zero
    _Float16 wq2t[32 * S_W2];     // Wq2^T  [e=32][j=264]
    _Float16 embo[KEYS * S_EO];   // emb_o  [key=256][e], e>=32 zero
    _Float16 wa1t[64 * S_A1];     // Wa1^T  [j=64][i], i>=32 zero
    _Float16 wa2t[16 * S_A2];     // Wa2 padded to 16 cols: row0 = Wa2, rest 0
    _Float16 inp [QTILE * S_INP]; // query MLP input [q=128][k=32]
    _Float16 hid [QTILE * S_INP]; // GEMM1 chunk (wave-private rows)
    _Float16 embq[QTILE * S_INP]; // emb_q [q=128][32] (wave-private rows)
    _Float16 hmask[QTILE * S_A2]; // mask-MLP hidden [q=128][64] (wave-private)
    _Float16 att [QTILE * S_ATT]; // scores -> exp() [q=128][256]
    _Float16 vbuf[CELL  * S_V];   // view_cell (Bt layout) [cell=128][256]
    float bq1[256];
    float bq2[32];
    float ba1[64];
    float ba2v;
    float mlogit[QTILE];          // mask logits per query row
    float red[256];               // per-half row max
    float red2[256];              // per-half row sum(exp)
};
static_assert(sizeof(Smem) <= 320 * 1024, "LDS over 320KB/WGP");

// --- fragment loaders matching the CDNA5 16-bit WMMA VGPR layouts ----------
// A (16xK row-major in LDS): lane l holds row M=l&15; halves 0..7 = K of
// (l>>4)*8 .., halves 8..15 = K of 16+(l>>4)*8 ..
__device__ __forceinline__ v16h load_a16(const _Float16* buf, int m0, int k0,
                                         int stride) {
    const int lane = threadIdx.x & 31;
    const _Float16* p =
        buf + (m0 + (lane & 15)) * stride + k0 + ((lane >> 4) << 3);
    union { v16h v; v8h h[2]; } u;
    u.h[0] = *(const v8h*)(p);
    u.h[1] = *(const v8h*)(p + 16);
    return u.v;
}
// B (Kx16, stored transposed: Bt[n][k] row-major): lane l holds col N=l&15,
// halves 0..7 = K of (l>>4)*8 .., halves 8..15 = K of 16+(l>>4)*8 ..
__device__ __forceinline__ v16h load_b16(const _Float16* bt, int n0, int k0,
                                         int stride) {
    const int lane = threadIdx.x & 31;
    const _Float16* p =
        bt + (n0 + (lane & 15)) * stride + k0 + ((lane >> 4) << 3);
    union { v16h v; v8h h[2]; } u;
    u.h[0] = *(const v8h*)(p);
    u.h[1] = *(const v8h*)(p + 16);
    return u.v;
}

#define WMMA_F16(A, B, C)                                                    \
    __builtin_amdgcn_wmma_f32_16x16x32_f16(false, (A), false, (B), (short)0, \
                                           (C), false, false)

// ---------------------------------------------------------------------------
// Kernel 1: quaternions + rigid relative pose  inv([Ro|to]) @ [Rq|tq]
// ---------------------------------------------------------------------------
__device__ __forceinline__ void mat2quat(const float* m, float* o7) {
    float m00 = m[0], m01 = m[1], m02 = m[2],  tx = m[3];
    float m10 = m[4], m11 = m[5], m12 = m[6],  ty = m[7];
    float m20 = m[8], m21 = m[9], m22 = m[10], tz = m[11];
    float t0 = 1.f + m00 - m11 - m22, t1 = 1.f - m00 + m11 - m22;
    float t2 = 1.f - m00 - m11 + m22, t3 = 1.f + m00 + m11 + m22;
    float t, q0, q1, q2, q3;
    if (m22 < 0.f) {
        if (m00 > m11) { t = t0; q0 = t0;        q1 = m01 + m10; q2 = m20 + m02; q3 = m12 - m21; }
        else           { t = t1; q0 = m01 + m10; q1 = t1;        q2 = m12 + m21; q3 = m20 - m02; }
    } else {
        if (m00 < -m11){ t = t2; q0 = m20 + m02; q1 = m12 + m21; q2 = t2;        q3 = m01 - m10; }
        else           { t = t3; q0 = m12 - m21; q1 = m20 - m02; q2 = m01 - m10; q3 = t3; }
    }
    float s = 0.5f * rsqrtf(t);
    o7[0] = tx; o7[1] = ty; o7[2] = tz;
    o7[3] = q0 * s; o7[4] = q1 * s; o7[5] = q2 * s; o7[6] = q3 * s;
}

__global__ void pose_kernel(const float* __restrict__ pose_o,
                            const float* __restrict__ pose_q,
                            float* __restrict__ out,      // full d_out
                            float* __restrict__ ptrans) { // ws [256][12]
    const int t = threadIdx.x;
    if (t < BB * NN) {
        const float* po = pose_o + t * 12;
        const float* pq = pose_q + (t >> 3) * 12;
        float q7[7];
        mat2quat(po, q7);
        float* dq = out + QUAT_O_OFF + t * 7;
        for (int i = 0; i < 7; ++i) dq[i] = q7[i];
        // rigid inverse compose: Rr = Ro^T Rq ; tr = Ro^T (tq - to)
        float o12[12];
        for (int i = 0; i < 3; ++i) {
            for (int j = 0; j < 3; ++j) {
                float s = 0.f;
                for (int k = 0; k < 3; ++k) s += po[k * 4 + i] * pq[k * 4 + j];
                o12[i * 4 + j] = s;
            }
            float s = 0.f;
            for (int k = 0; k < 3; ++k)
                s += po[k * 4 + i] * (pq[k * 4 + 3] - po[k * 4 + 3]);
            o12[i * 4 + 3] = s;
        }
        for (int i = 0; i < 12; ++i) ptrans[t * 12 + i] = o12[i];
    } else if (t < BB * NN + BB) {
        const int bb = t - BB * NN;
        float q7[7];
        mat2quat(pose_q + bb * 12, q7);
        float* dq = out + QUAT_Q_OFF + bb * 7;
        for (int i = 0; i < 7; ++i) dq[i] = q7[i];
    }
}

// ---------------------------------------------------------------------------
// Kernel 2: fused  query-MLP -> mask-MLP -> attention -> att@v -> N-sum
// grid = B * NQT workgroups; 256 threads; wave w owns query rows w*16..+15.
// ---------------------------------------------------------------------------
__global__ __launch_bounds__(256, 1) void fused_kernel(
    const float* __restrict__ view_cell,
    const float* __restrict__ Wk1, const float* __restrict__ bk1,
    const float* __restrict__ Wk2, const float* __restrict__ bk2,
    const float* __restrict__ Wq1, const float* __restrict__ bq1g,
    const float* __restrict__ Wq2, const float* __restrict__ bq2g,
    const float* __restrict__ Wa1, const float* __restrict__ ba1g,
    const float* __restrict__ Wa2, const float* __restrict__ ba2g,
    const float* __restrict__ ptrans, float* __restrict__ out) {
    __shared__ Smem s;
    const int tid  = threadIdx.x;
    const int lane = tid & 31;
    const int m0   = (tid >> 5) * 16;       // wave's first query row
    const int b    = blockIdx.x / NQT;
    const int q0   = (blockIdx.x % NQT) * QTILE;
    const int ncol = lane & 15;             // C-tile column held by this lane
    const int mb   = (lane >> 4) * 8;       // C-tile row base held by this lane

    // ---- one-time preload: transposed weights, biases, key embeddings ----
    {
        const int j = tid;                                   // 0..255
        for (int k = 0; k < S_W1; ++k)
            s.wq1t[j * S_W1 + k] =
                (k < 15) ? (_Float16)Wq1[k * 256 + j] : (_Float16)0.f;
        s.bq1[j] = bq1g[j];
    }
    for (int idx = tid; idx < 32 * S_W2; idx += 256) {
        const int e = idx / S_W2, jj = idx % S_W2;
        s.wq2t[idx] = (jj < 256) ? (_Float16)Wq2[jj * EMB + e] : (_Float16)0.f;
    }
    if (tid < 32) s.bq2[tid] = bq2g[tid];
    if (tid < 64) {
        for (int i = 0; i < S_A1; ++i)
            s.wa1t[tid * S_A1 + i] =
                (i < EMB) ? (_Float16)Wa1[i * 64 + tid] : (_Float16)0.f;
        s.ba1[tid] = ba1g[tid];
    }
    if (tid < 16) {
        for (int j = 0; j < S_A2; ++j)
            s.wa2t[tid * S_A2 + j] =
                (tid == 0 && j < 64) ? (_Float16)Wa2[j] : (_Float16)0.f;
    }
    if (tid == 0) s.ba2v = ba2g[0];
    {   // emb_o = MLP(code_o): one key per thread (weights are scalar-uniform)
        const int key = tid;
        const float hg = -1.f + 2.f * (float)(key >> 4) / 15.f;
        const float wg = -1.f + 2.f * (float)(key & 15) / 15.f;
        float acc[EMB];
        for (int e = 0; e < EMB; ++e) acc[e] = bk2[e];
        for (int i = 0; i < 128; ++i) {
            const float hv = fmaxf(hg * Wk1[i] + wg * Wk1[128 + i] + bk1[i], 0.f);
            for (int e = 0; e < EMB; ++e) acc[e] += hv * Wk2[i * EMB + e];
        }
        for (int e = 0; e < EMB; ++e) s.embo[key * S_EO + e] = (_Float16)acc[e];
        for (int e = EMB; e < S_EO; ++e) s.embo[key * S_EO + e] = (_Float16)0.f;
    }

    // persistent accumulator: sum over n of mask*softmax@v  [16q x 128cell]
    v8f acc[8];
    for (int ct = 0; ct < 8; ++ct) { v8f z = {}; acc[ct] = z; }

    __syncthreads();

    for (int n = 0; n < NN; ++n) {
        const int bn = b * NN + n;

        // ---- build inp rows [128 x 32] : 12 pose + 3 code + zero pad ----
        if (tid < QTILE) {
            const int q = q0 + tid;
            const int d = q >> 8, rem = q & 255;
            const float* pt = ptrans + bn * 12;   // uniform -> s_loads
            _Float16* row = s.inp + tid * S_INP;
            for (int k = 0; k < 12; ++k) row[k] = (_Float16)pt[k];
            row[12] = (_Float16)((float)d / 5.f);
            row[13] = (_Float16)(-1.f + 2.f * (float)(rem >> 4) / 15.f);
            row[14] = (_Float16)(-1.f + 2.f * (float)(rem & 15) / 15.f);
            for (int k = 15; k < 32; ++k) row[k] = (_Float16)0.f;
        }
        __syncthreads();   // B1: inp visible (only cross-wave producer here)

        // ---- GEMM1 (inp x Wq1) fused with GEMM2 (hid x Wq2), K-chunked.
        // hid rows are wave-private; LDS is in-order per wave => no barriers.
        const v16h a_inp = load_a16(s.inp, m0, 0, S_INP);
        v8f emb0 = {}, emb1 = {};
        for (int c = 0; c < 8; ++c) {
            for (int t = 0; t < 2; ++t) {
                const int j0 = c * 32 + t * 16;
                const v16h bf = load_b16(s.wq1t, j0, 0, S_W1);
                v8f z = {};
                const v8f hf = WMMA_F16(a_inp, bf, z);
                const float bj = s.bq1[j0 + ncol];
                for (int r = 0; r < 8; ++r) {
                    const float hv = fmaxf(hf[r] + bj, 0.f);
                    s.hid[(m0 + mb + r) * S_INP + t * 16 + ncol] = (_Float16)hv;
                }
            }
            const v16h a_h = load_a16(s.hid, m0, 0, S_INP);
            emb0 = WMMA_F16(a_h, load_b16(s.wq2t, 0,  c * 32, S_W2), emb0);
            emb1 = WMMA_F16(a_h, load_b16(s.wq2t, 16, c * 32, S_W2), emb1);
        }
        {   // + bq2, stash emb_q (wave-private rows)
            const float b0v = s.bq2[ncol], b1v = s.bq2[16 + ncol];
            for (int r = 0; r < 8; ++r) {
                s.embq[(m0 + mb + r) * S_INP + ncol]      = (_Float16)(emb0[r] + b0v);
                s.embq[(m0 + mb + r) * S_INP + 16 + ncol] = (_Float16)(emb1[r] + b1v);
            }
        }
        const v16h a_e = load_a16(s.embq, m0, 0, S_INP);

        // ---- mask MLP on the matrix pipe (reuses a_e) ----
        // hidden = relu(emb_q @ Wa1 + ba1)   [16 x 64]
        for (int t = 0; t < 4; ++t) {
            v8f z = {};
            const v8f hf = WMMA_F16(a_e, load_b16(s.wa1t, t * 16, 0, S_A1), z);
            const float bj = s.ba1[t * 16 + ncol];
            for (int r = 0; r < 8; ++r)
                s.hmask[(m0 + mb + r) * S_A2 + t * 16 + ncol] =
                    (_Float16)fmaxf(hf[r] + bj, 0.f);
        }
        {   // logit = hidden @ Wa2 (padded to 16 cols; col 0 is real)
            v8f z = {};
            v8f m2 = WMMA_F16(load_a16(s.hmask, m0, 0, S_A2),
                              load_b16(s.wa2t, 0, 0, S_A2), z);
            m2 = WMMA_F16(load_a16(s.hmask, m0, 32, S_A2),
                          load_b16(s.wa2t, 0, 32, S_A2), m2);
            if (ncol == 0)   // lanes 0 & 16 hold column 0 of the C tile
                for (int r = 0; r < 8; ++r)
                    s.mlogit[m0 + mb + r] = m2[r] + s.ba2v;
        }

        // ---- scores = emb_q @ emb_o^T  -> raw f16 into s.att ----
        for (int kt = 0; kt < 16; ++kt) {
            const v16h b_o = load_b16(s.embo, kt * 16, 0, S_EO);
            v8f z = {};
            const v8f sf = WMMA_F16(a_e, b_o, z);
            for (int r = 0; r < 8; ++r)
                s.att[(m0 + mb + r) * S_ATT + kt * 16 + ncol] = (_Float16)sf[r];
        }

        // ---- stage view_cell[bn] (already [cell][hw] = Bt layout) as f16 ----
        {
            const float* vsrc = view_cell + (size_t)bn * (CELL * KEYS);
            for (int i = 0; i < 32; ++i) {
                const int f = i * 1024 + tid * 4;
                const float4 vv = *(const float4*)(vsrc + f);
                _Float16* dst = s.vbuf + (f >> 8) * S_V + (f & 255);
                dst[0] = (_Float16)vv.x; dst[1] = (_Float16)vv.y;
                dst[2] = (_Float16)vv.z; dst[3] = (_Float16)vv.w;
            }
            if (n + 1 < NN)   // hide next slab's L2 latency (global_prefetch_b8)
                __builtin_prefetch(view_cell + (size_t)(bn + 1) * (CELL * KEYS)
                                   + tid * 128, 0, 1);
        }

        // ---- softmax stats, 2 threads per row; the producing threads are
        // the row-owning wave itself, so only same-wave LDS ordering is used.
        {
            const int r2 = tid >> 1, hf2 = tid & 1;
            const _Float16* arow = s.att + r2 * S_ATT + hf2 * 128;
            float mx = -1e30f;
            for (int k = 0; k < 128; ++k) mx = fmaxf(mx, (float)arow[k]);
            s.red[tid] = mx;
        }
        {
            const int r2 = tid >> 1, hf2 = tid & 1;
            const float mx = fmaxf(s.red[tid & ~1], s.red[tid | 1]);
            _Float16* arow = s.att + r2 * S_ATT + hf2 * 128;
            float sum = 0.f;
            for (int k = 0; k < 128; ++k) {
                const float ev = __expf((float)arow[k] - mx);
                sum += ev;
                arow[k] = (_Float16)ev;   // write back exp(s - max)
            }
            s.red2[tid] = sum;
        }
        __syncthreads();   // Bv: vbuf (cross-wave) ready; att/red2 already are

        // ---- vatt: (mask*softmax) @ v, accumulated across n in registers.
        // A-fragment lanes each hold one M row -> per-lane scalar scale,
        // applied as packed f16 multiplies (v_pk_mul_f16).
        {
            const int mrow = m0 + (lane & 15);
            const float mask = 1.f / (1.f + __expf(-s.mlogit[mrow]));
            const float sc = mask / (s.red2[2 * mrow] + s.red2[2 * mrow + 1]);
            const _Float16 sch = (_Float16)sc;
            for (int kt = 0; kt < 8; ++kt) {
                const v16h a_raw = load_a16(s.att, m0, kt * 32, S_ATT);
                const v16h a_s = a_raw * sch;
                for (int ct = 0; ct < 8; ++ct) {
                    const v16h b_v = load_b16(s.vbuf, ct * 16, kt * 32, S_V);
                    acc[ct] = WMMA_F16(a_s, b_v, acc[ct]);
                }
            }
        }
        __syncthreads();   // B4: all vatt reads done before next n rewrites
    }

    // ---- epilogue: sigmoid, write out[b][cell][q] ----
    for (int ct = 0; ct < 8; ++ct) {
        const int cell = ct * 16 + ncol;
        for (int r = 0; r < 8; ++r) {
            const int q = q0 + m0 + mb + r;
            const float v = 1.f / (1.f + __expf(-acc[ct][r]));
            out[((size_t)b * CELL + cell) * QTOT + q] = v;
        }
    }
}

// ---------------------------------------------------------------------------
extern "C" void kernel_launch(void* const* d_in, const int* in_sizes, int n_in,
                              void* d_out, int out_size, void* d_ws,
                              size_t ws_size, hipStream_t stream) {
    const float* view_cell = (const float*)d_in[0];
    const float* pose_o    = (const float*)d_in[1];
    const float* pose_q    = (const float*)d_in[2];
    const float* Wk1 = (const float*)d_in[3];
    const float* bk1 = (const float*)d_in[4];
    const float* Wk2 = (const float*)d_in[5];
    const float* bk2 = (const float*)d_in[6];
    const float* Wq1 = (const float*)d_in[7];
    const float* bq1 = (const float*)d_in[8];
    const float* Wq2 = (const float*)d_in[9];
    const float* bq2 = (const float*)d_in[10];
    const float* Wa1 = (const float*)d_in[11];
    const float* ba1 = (const float*)d_in[12];
    const float* Wa2 = (const float*)d_in[13];
    const float* ba2 = (const float*)d_in[14];
    float* out    = (float*)d_out;
    float* ptrans = (float*)d_ws;   // 256*12 floats

    pose_kernel<<<1, 320, 0, stream>>>(pose_o, pose_q, out, ptrans);
    fused_kernel<<<BB * NQT, 256, 0, stream>>>(
        view_cell, Wk1, bk1, Wk2, bk2, Wq1, bq1, Wq2, bq2, Wa1, ba1, Wa2, ba2,
        ptrans, out);
}
